// CoAtt_11579231830661
// MI455X (gfx1250) — matrix operand order repaired and checked by reading, compiled, not run on hardware
//
#include <hip/hip_runtime.h>

typedef __attribute__((ext_vector_type(16))) _Float16 v16h;
typedef __attribute__((ext_vector_type(8)))  _Float16 v8h;
typedef __attribute__((ext_vector_type(8)))  float    v8f;

#define BB 8
#define CC 64
#define WW 2048
#define NTOT (BB*CC*WW)   // 1,048,576

// ---------------------------------------------------------------------------
// Fragment builders (ISA 7.12.2 layouts, wave32)
//  A 16-bit 16x32: lanes 0-15 hold K 0..7 & 16..23, lanes 16-31 hold K 8..15 & 24..31
//    -> p points at (row base + kchunk + (hi?8:0)); two b128 loads at +0, +16
//  B 16-bit 32x16: lanes 0-15 hold K 0..7 & 16..23, lanes 16-31 hold K 8..15 & 24..31
//    -> p points at (col base + kchunk + (hi?8:0)); two b128 loads at +0, +16
// ---------------------------------------------------------------------------
static __device__ __forceinline__ v16h load_a_frag(const _Float16* p) {
  v8h lo = *(const v8h*)(p);
  v8h hi = *(const v8h*)(p + 16);
  return __builtin_shufflevector(lo, hi, 0,1,2,3,4,5,6,7,8,9,10,11,12,13,14,15);
}
static __device__ __forceinline__ v16h load_b_frag(const _Float16* p) {
  v8h lo = *(const v8h*)(p);
  v8h hi = *(const v8h*)(p + 16);
  return __builtin_shufflevector(lo, hi, 0,1,2,3,4,5,6,7,8,9,10,11,12,13,14,15);
}

// XOR butterfly reductions via ds_swizzle (group-of-32: offset = (xor<<10)|0x1f)
static __device__ __forceinline__ float wave_xor_max(float v) {
  v = fmaxf(v, __int_as_float(__builtin_amdgcn_ds_swizzle(__float_as_int(v), 0x401f)));
  v = fmaxf(v, __int_as_float(__builtin_amdgcn_ds_swizzle(__float_as_int(v), 0x201f)));
  v = fmaxf(v, __int_as_float(__builtin_amdgcn_ds_swizzle(__float_as_int(v), 0x101f)));
  v = fmaxf(v, __int_as_float(__builtin_amdgcn_ds_swizzle(__float_as_int(v), 0x081f)));
  v = fmaxf(v, __int_as_float(__builtin_amdgcn_ds_swizzle(__float_as_int(v), 0x041f)));
  return v;
}
static __device__ __forceinline__ float wave_xor_sum(float v) {
  v += __int_as_float(__builtin_amdgcn_ds_swizzle(__float_as_int(v), 0x401f));
  v += __int_as_float(__builtin_amdgcn_ds_swizzle(__float_as_int(v), 0x201f));
  v += __int_as_float(__builtin_amdgcn_ds_swizzle(__float_as_int(v), 0x101f));
  v += __int_as_float(__builtin_amdgcn_ds_swizzle(__float_as_int(v), 0x081f));
  v += __int_as_float(__builtin_amdgcn_ds_swizzle(__float_as_int(v), 0x041f));
  return v;
}

// ---------------------------------------------------------------------------
// Kernel 1: per-block partial sum / sum-of-squares over x
// ---------------------------------------------------------------------------
__global__ __launch_bounds__(256) void stats_partial(const float* __restrict__ x,
                                                     float* __restrict__ partials) {
  __shared__ float s1[256], s2[256];
  int tid = threadIdx.x;
  float s = 0.f, ss = 0.f;
  for (size_t i = (size_t)blockIdx.x * 256 + tid; i < (size_t)NTOT; i += 262144) {
    float v = x[i];
    s += v; ss += v * v;
  }
  s1[tid] = s; s2[tid] = ss;
  __syncthreads();
  for (int off = 128; off > 0; off >>= 1) {
    if (tid < off) { s1[tid] += s1[tid + off]; s2[tid] += s2[tid + off]; }
    __syncthreads();
  }
  if (tid == 0) { partials[2*blockIdx.x] = s1[0]; partials[2*blockIdx.x + 1] = s2[0]; }
}

// ---------------------------------------------------------------------------
// Kernel 2: finalize mean and 1/std (unbiased, ddof=1)
// ---------------------------------------------------------------------------
__global__ __launch_bounds__(256) void stats_final(const float* __restrict__ partials,
                                                   float* __restrict__ stats) {
  __shared__ float s1[256], s2[256];
  int tid = threadIdx.x;
  float s = 0.f, ss = 0.f;
  for (int i = tid; i < 1024; i += 256) { s += partials[2*i]; ss += partials[2*i + 1]; }
  s1[tid] = s; s2[tid] = ss;
  __syncthreads();
  for (int off = 128; off > 0; off >>= 1) {
    if (tid < off) { s1[tid] += s1[tid + off]; s2[tid] += s2[tid + off]; }
    __syncthreads();
  }
  if (tid == 0) {
    const float N = (float)NTOT;
    float mean = s1[0] / N;
    float var  = (s2[0] - s1[0] * s1[0] / N) / (N - 1.0f);
    stats[0] = mean;
    stats[1] = rsqrtf(var);
  }
}

// ---------------------------------------------------------------------------
// Kernel 3: per-batch Gram matrix  G = xz @ xz^T  (64x64)
//   G[c1,c2] = inv_std^2 * ( S[c1,c2] - mean*(R[c1]+R[c2]) + W*mean^2 )
// ---------------------------------------------------------------------------
__global__ __launch_bounds__(256) void gram_kernel(const float* __restrict__ x,
                                                   const float* __restrict__ stats,
                                                   float* __restrict__ G) {
  __shared__ float xs[64 * 128];   // 32 KB chunk of x[b]
  __shared__ float rs[64];
  int b = blockIdx.x, tid = threadIdx.x;
  int c1  = tid >> 2;
  int c2b = (tid & 3) * 16;
  float acc[16];
#pragma unroll
  for (int j = 0; j < 16; ++j) acc[j] = 0.f;
  float rsacc = 0.f;

  for (int ch = 0; ch < 16; ++ch) {
    for (int e = tid; e < 8192; e += 256) {
      int c = e >> 7, w = e & 127;
      xs[e] = x[((size_t)b * CC + c) * WW + ch * 128 + w];
    }
    __syncthreads();
    for (int w = 0; w < 128; ++w) {
      float a = xs[c1 * 128 + w];
#pragma unroll
      for (int j = 0; j < 16; ++j) acc[j] += a * xs[(c2b + j) * 128 + w];
    }
    if (tid < 64) {
      float r = 0.f;
      for (int w = 0; w < 128; ++w) r += xs[tid * 128 + w];
      rsacc += r;
    }
    __syncthreads();
  }
  if (tid < 64) rs[tid] = rsacc;
  __syncthreads();

  float mean = stats[0], is = stats[1];
  float is2 = is * is;
#pragma unroll
  for (int j = 0; j < 16; ++j) {
    int c2 = c2b + j;
    G[(size_t)b * 4096 + c1 * 64 + c2] =
        is2 * (acc[j] - mean * (rs[c1] + rs[c2]) + (float)WW * mean * mean);
  }
}

// ---------------------------------------------------------------------------
// Kernel 4: projections pq/pk/pv = W{q,k,v} @ x + b  (f32 math), fused T = G@pk.
// Outputs (all WMMA-feed-ready, rows contiguous per lane):
//   pqT : [B][W][C] f16   (energy-GEMM A rows)
//   pvh : [B][C][W] f16   (out-GEMM A rows)
//   TT  : [B][W][C] f16   (energy-GEMM B columns, w-major)
// ---------------------------------------------------------------------------
__global__ __launch_bounds__(256) void proj_kernel(
    const float* __restrict__ x,
    const float* __restrict__ Wq, const float* __restrict__ bq,
    const float* __restrict__ Wk, const float* __restrict__ bk,
    const float* __restrict__ Wv, const float* __restrict__ bv,
    const float* __restrict__ G,
    _Float16* __restrict__ pqT, _Float16* __restrict__ pvh, _Float16* __restrict__ TT) {
  __shared__ float wq[64 * 64], wk[64 * 64], wv_[64 * 64];  // 48 KB
  __shared__ float xc[64 * 32];                             // 8 KB
  __shared__ float pkl[64 * 32];                            // 8 KB (total 64 KB)
  int b = blockIdx.x, w0 = blockIdx.y * 32, tid = threadIdx.x;

  for (int i = tid; i < 4096; i += 256) { wq[i] = Wq[i]; wk[i] = Wk[i]; wv_[i] = Wv[i]; }
  for (int i = tid; i < 2048; i += 256) {
    int c = i >> 5, w = i & 31;
    xc[i] = x[((size_t)b * CC + c) * WW + w0 + w];
  }
  __syncthreads();

  for (int j = 0; j < 8; ++j) {
    int idx = tid + 256 * j;
    int o = idx >> 5, w = idx & 31;
    float sq = bq[o], sk = bk[o], sv = bv[o];
    for (int c = 0; c < 64; ++c) {
      float xv = xc[c * 32 + w];
      sq += wq[o * 64 + c] * xv;
      sk += wk[o * 64 + c] * xv;
      sv += wv_[o * 64 + c] * xv;
    }
    pqT[((size_t)b * WW + w0 + w) * CC + o] = (_Float16)sq;
    pvh[((size_t)b * CC + o) * WW + w0 + w] = (_Float16)sv;
    pkl[o * 32 + w] = sk;
  }
  __syncthreads();

  for (int j = 0; j < 8; ++j) {
    int idx = tid + 256 * j;
    int o = idx >> 5, w = idx & 31;
    const float* g = G + (size_t)b * 4096 + o * 64;
    float s = 0.f;
    for (int c = 0; c < 64; ++c) s += g[c] * pkl[c * 32 + w];
    TT[((size_t)b * WW + w0 + w) * CC + o] = (_Float16)s;   // w-major
  }
}

// ---------------------------------------------------------------------------
// Kernel 5 (fused attention): per (batch, 16-row v-block):
//   E[v,:]  = pqT_blk(16x64) @ T(64x2048)     -- v_wmma_f32_16x16x32_f16, K=64
//   att     = softmax(E, axis=w)  in-place in 64 KB LDS (b128 vector passes)
//   out_blk = pv(64x2048) @ att^T             -- WMMA, K=2048
//   out     = gamma*out_blk + (x-mean)*inv_std
// Grid: (B, W/16), 128 threads (4 waves, wave32).
// ---------------------------------------------------------------------------
__global__ __launch_bounds__(128) void attn_kernel(
    const float* __restrict__ x,
    const _Float16* __restrict__ pqT, const _Float16* __restrict__ pvh,
    const _Float16* __restrict__ TT,
    const float* __restrict__ stats, const float* __restrict__ gamma,
    float* __restrict__ out) {
  __shared__ _Float16 E[16 * 2048];   // exactly 64 KB

  int b = blockIdx.x, v0 = blockIdx.y * 16;
  int tid = threadIdx.x, lane = tid & 31, wv = tid >> 5;
  int ln = lane & 15;
  int hi = (lane >= 16) ? 1 : 0;
  int akb = hi ? 8 : 0;    // A half-lane K base (interleaved layout)
  int bkb = hi ? 8 : 0;    // B half-lane K base (interleaved layout)

  // ---- A fragments: pqT rows (m = v0+ln), two K-chunks of 32 channels ----
  const _Float16* arow = pqT + ((size_t)b * WW + v0 + ln) * CC;
  v16h aq0 = load_a_frag(arow + akb);
  v16h aq1 = load_a_frag(arow + 32 + akb);

  // ---- Energy tiles: wave wv covers w-tiles wv, wv+4, ... ----
  for (int tile = wv; tile < 128; tile += 4) {
    int w0 = tile * 16;
    const _Float16* bcol = TT + ((size_t)b * WW + w0 + ln) * CC;
    v8f acc = {};
    v16h bf0 = load_b_frag(bcol + bkb);
    v16h bf1 = load_b_frag(bcol + 32 + bkb);
    acc = __builtin_amdgcn_wmma_f32_16x16x32_f16(false, aq0, false, bf0,
                                                 (short)0, acc, false, false);
    acc = __builtin_amdgcn_wmma_f32_16x16x32_f16(false, aq1, false, bf1,
                                                 (short)0, acc, false, false);
#pragma unroll
    for (int r = 0; r < 8; ++r) {
      int row = hi ? (8 + r) : r;   // D: VGPR r -> M=r (lanes 0-15) / M=8+r (16-31)
      E[row * 2048 + w0 + ln] = (_Float16)acc[r];
    }
  }
  __syncthreads();

  // ---- Softmax over w: wave wv owns rows wv..; b128 LDS accesses ----
  for (int row = wv; row < 16; row += 4) {
    v8h* erow = (v8h*)&E[row * 2048];   // 256 chunks of 8 halves
    float mx = -3.402823466e+38f;
    for (int j = lane; j < 256; j += 32) {
      v8h e = erow[j];
#pragma unroll
      for (int i = 0; i < 8; ++i) mx = fmaxf(mx, (float)e[i]);
    }
    mx = wave_xor_max(mx);
    float s = 0.f;
    for (int j = lane; j < 256; j += 32) {
      v8h e = erow[j];
#pragma unroll
      for (int i = 0; i < 8; ++i) s += __expf((float)e[i] - mx);
    }
    s = wave_xor_sum(s);
    float inv = 1.0f / s;
    for (int j = lane; j < 256; j += 32) {
      v8h e = erow[j];
      v8h o;
#pragma unroll
      for (int i = 0; i < 8; ++i) o[i] = (_Float16)(__expf((float)e[i] - mx) * inv);
      erow[j] = o;
    }
  }
  __syncthreads();

  // ---- out tile: c in [wv*16, wv*16+16), v in [v0, v0+16), K = 2048 ----
  float mean = stats[0], inv_std = stats[1], g = gamma[0];
  v8f acc = {};
  const _Float16* arow2 = pvh + ((size_t)b * CC + wv * 16 + ln) * WW;
  const _Float16* brow  = &E[ln * 2048];
#pragma unroll 4
  for (int w0 = 0; w0 < 2048; w0 += 32) {
    v16h af = load_a_frag(arow2 + w0 + akb);
    v16h bf = load_b_frag(brow + w0 + bkb);
    acc = __builtin_amdgcn_wmma_f32_16x16x32_f16(false, af, false, bf,
                                                 (short)0, acc, false, false);
  }
#pragma unroll
  for (int r = 0; r < 8; ++r) {
    int c = wv * 16 + (hi ? (8 + r) : r);
    size_t oi = ((size_t)b * CC + c) * WW + v0 + ln;
    float xz = (x[oi] - mean) * inv_std;
    out[oi] = g * acc[r] + xz;
  }
}

// ---------------------------------------------------------------------------
// Launch
// ---------------------------------------------------------------------------
extern "C" void kernel_launch(void* const* d_in, const int* in_sizes, int n_in,
                              void* d_out, int out_size, void* d_ws, size_t ws_size,
                              hipStream_t stream) {
  (void)in_sizes; (void)n_in; (void)out_size; (void)ws_size;
  const float* x     = (const float*)d_in[0];
  const float* Wq    = (const float*)d_in[1];
  const float* bq    = (const float*)d_in[2];
  const float* Wk    = (const float*)d_in[3];
  const float* bk    = (const float*)d_in[4];
  const float* Wv    = (const float*)d_in[5];
  const float* bv    = (const float*)d_in[6];
  const float* gamma = (const float*)d_in[7];
  float* out = (float*)d_out;

  // workspace: [partials 2048 f32][stats 16 f32][G 32768 f32][3 x 1M f16]
  float* wsf      = (float*)d_ws;
  float* partials = wsf;
  float* stats    = wsf + 2048;
  float* G        = wsf + 2064;
  _Float16* pqT = (_Float16*)(wsf + 2064 + BB * CC * CC);  // [B][W][C]
  _Float16* pvh = pqT + (size_t)BB * WW * CC;              // [B][C][W]
  _Float16* TT  = pvh + (size_t)BB * WW * CC;              // [B][W][C]

  stats_partial<<<1024, 256, 0, stream>>>(x, partials);
  stats_final<<<1, 256, 0, stream>>>(partials, stats);
  gram_kernel<<<BB, 256, 0, stream>>>(x, stats, G);
  proj_kernel<<<dim3(BB, WW / 32), 256, 0, stream>>>(x, Wq, bq, Wk, bk, Wv, bv, G,
                                                     pqT, pvh, TT);
  attn_kernel<<<dim3(BB, WW / 16), 128, 0, stream>>>(x, pqT, pvh, TT, stats, gamma, out);
}
